// DecoderAttentionLSTM_89026082112160
// MI455X (gfx1250) — matrix operand order repaired and checked by reading, not compiled, not run on hardware
//
#include <hip/hip_runtime.h>
#include <hip/hip_bf16.h>

// Problem sizes (fixed by the reference).
constexpr int VOC = 32000;
constexpr int HD  = 1024;
constexpr int SEQ = 512;
constexpr int TT  = 256;

typedef __attribute__((ext_vector_type(16))) __bf16 v16bf;
typedef __attribute__((ext_vector_type(8)))  __bf16 v8bf;
typedef __attribute__((ext_vector_type(8)))  float  v8f;

// ---------------------------------------------------------------------------
// wave32 reductions
// ---------------------------------------------------------------------------
__device__ __forceinline__ float wave_sum(float v) {
#pragma unroll
  for (int off = 16; off > 0; off >>= 1) v += __shfl_down(v, off, 32);
  return v;
}
__device__ __forceinline__ float wave_max(float v) {
#pragma unroll
  for (int off = 16; off > 0; off >>= 1) v = fmaxf(v, __shfl_down(v, off, 32));
  return v;
}

// ---------------------------------------------------------------------------
// Gather token embeddings (teacher forcing: tokens known up front) -> bf16
// ---------------------------------------------------------------------------
__global__ void gather_embed_bf16(const int* __restrict__ start_token,
                                  const int* __restrict__ target,
                                  const float* __restrict__ embed,
                                  __bf16* __restrict__ embA) {
  int t = blockIdx.x;                         // 0..TT-1
  int tok = (t == 0) ? start_token[0] : target[t - 1];
  const float* src = embed + (size_t)tok * HD;
  for (int i = threadIdx.x; i < HD; i += blockDim.x)
    embA[(size_t)t * HD + i] = (__bf16)src[i];
}

__global__ void f32_to_bf16(const float* __restrict__ src,
                            __bf16* __restrict__ dst, int n) {
  int i = blockIdx.x * blockDim.x + threadIdx.x;
  if (i < n) dst[i] = (__bf16)src[i];
}

// ---------------------------------------------------------------------------
// WMMA bf16 GEMM:  D[M,N] = A[M,K](bf16) * Bw[N,K]^T (f32, row stride ldb,
// converted to bf16 on the fly) + bias0[n] + bias1[n].
//
// Block = 8 waves; wave w computes a 16x32 strip (2 accumulators) at
// (blockIdx.y*16, blockIdx.x*256 + w*32).  One A fragment feeds two
// v_wmma_f32_16x16x32_bf16 per K-step; K-loop unrolled x2 so 4 independent
// WMMAs are in flight (hides the WMMA RAW latency, ISA 7.12.1).
//
// VGPR layouts used here (A and B share contiguous K order):
//   A 16x32: lane 0-15 = row M; lane-half 0 covers K {0..15},
//            half 1 covers {16..31} -> two contiguous 16B loads.
//   B 32x16: VGPR r lo/hi = rows K=2r,2r+1 (+16 for upper half), col=lane&15
//            -> 16 contiguous fp32 per lane, cvt_pk to bf16.
//   D f32 16x16: VGPR r -> M = r + 8*(lane>>4), N = lane & 15.
// ---------------------------------------------------------------------------
__device__ __forceinline__ v16bf load_a_frag(const __bf16* Arow, int kb,
                                             int half) {
  const v8bf lo = *(const v8bf*)(Arow + kb + 16 * half);       // K run 1
  const v8bf hi = *(const v8bf*)(Arow + kb + 16 * half + 8);   // K run 2
  return __builtin_shufflevector(lo, hi, 0, 1, 2, 3, 4, 5, 6, 7,
                                 8, 9, 10, 11, 12, 13, 14, 15);
}

__device__ __forceinline__ v16bf load_b_frag(const float* Brow, int kb,
                                             int half) {
  const float4* bp = (const float4*)(Brow + kb + 16 * half);   // 64B / lane
  const float4 x0 = bp[0], x1 = bp[1], x2 = bp[2], x3 = bp[3];
  v16bf b;
  b[0]  = (__bf16)x0.x; b[1]  = (__bf16)x0.y;
  b[2]  = (__bf16)x0.z; b[3]  = (__bf16)x0.w;
  b[4]  = (__bf16)x1.x; b[5]  = (__bf16)x1.y;
  b[6]  = (__bf16)x1.z; b[7]  = (__bf16)x1.w;
  b[8]  = (__bf16)x2.x; b[9]  = (__bf16)x2.y;
  b[10] = (__bf16)x2.z; b[11] = (__bf16)x2.w;
  b[12] = (__bf16)x3.x; b[13] = (__bf16)x3.y;
  b[14] = (__bf16)x3.z; b[15] = (__bf16)x3.w;
  return b;
}

__global__ __launch_bounds__(256)
void gemm_wmma_bf16(const __bf16* __restrict__ A,
                    const float* __restrict__ Bw, int ldb,
                    const float* __restrict__ bias0,
                    const float* __restrict__ bias1,
                    float* __restrict__ D,
                    int M, int N, int K) {
  const int wv   = threadIdx.x >> 5;
  const int l    = threadIdx.x & 31;
  const int half = l >> 4;
  const int lr   = l & 15;
  const int n0 = blockIdx.x * 256 + wv * 32;   // wave strip: cols [n0, n0+32)
  const int m0 = blockIdx.y * 16;

  const __bf16* Arow  = A  + (size_t)(m0 + lr) * K;
  const float*  Brow0 = Bw + (size_t)(n0 + lr) * ldb;
  const float*  Brow1 = Bw + (size_t)(n0 + 16 + lr) * ldb;

  v8f acc0 = {}, acc1 = {};
#pragma unroll 2
  for (int kb = 0; kb < K; kb += 32) {
    if (kb + 32 < K) {  // global_prefetch_b8 of next weight tiles
      __builtin_prefetch(&Brow0[kb + 32 + 16 * half], 0, 1);
      __builtin_prefetch(&Brow1[kb + 32 + 16 * half], 0, 1);
    }
    const v16bf a  = load_a_frag(Arow, kb, half);
    const v16bf b0 = load_b_frag(Brow0, kb, half);
    const v16bf b1 = load_b_frag(Brow1, kb, half);
    acc0 = __builtin_amdgcn_wmma_f32_16x16x32_bf16(
        false, a, false, b0, (short)0, acc0, false, false);
    acc1 = __builtin_amdgcn_wmma_f32_16x16x32_bf16(
        false, a, false, b1, (short)0, acc1, false, false);
  }

  const int na = n0 + lr;
  const int nb = n0 + 16 + lr;
  float ba = 0.f, bb = 0.f;
  if (bias0) { ba += bias0[na]; bb += bias0[nb]; }
  if (bias1) { ba += bias1[na]; bb += bias1[nb]; }
#pragma unroll
  for (int r = 0; r < 8; ++r) {
    const int m = r + 8 * half;
    float* drow = D + (size_t)(m0 + m) * N;
    drow[na] = acc0[r] + ba;
    drow[nb] = acc1[r] + bb;
  }
}

// ---------------------------------------------------------------------------
// Sequential recurrence: one workgroup of 1024 threads (32 wave32s) walks the
// 256 time steps.  State (h, ctx, scores, gates) lives in LDS; dot products
// are wave-parallel with shuffle reductions.  emb_pre already contains
// W_ih[:, :H] @ emb_t + b_ih + b_hh.
// ---------------------------------------------------------------------------
__global__ __launch_bounds__(1024)
void decoder_recurrence(const float* __restrict__ h0,
                        const float* __restrict__ c0,
                        const float* __restrict__ enc,       // [SEQ,HD]
                        const float* __restrict__ enc_proj,  // [SEQ,HD]
                        const float* __restrict__ w_comb,    // [HD]
                        const float* __restrict__ W_hid,     // [HD,HD]
                        const float* __restrict__ W_ih,      // [4HD,2HD]
                        const float* __restrict__ W_hh,      // [4HD,HD]
                        const float* __restrict__ emb_pre,   // [TT,4HD]
                        __bf16* __restrict__ hA)             // [TT,HD] out
{
  __shared__ float h_s[HD];
  __shared__ float Wh[HD];
  __shared__ float ctx[HD];
  __shared__ float sc[SEQ];
  __shared__ float gates[4 * HD];
  __shared__ float red[32];

  const int tid = threadIdx.x;
  const int l   = tid & 31;
  const int wv  = tid >> 5;

  float c = c0[tid];
  h_s[tid] = h0[tid];
  __syncthreads();

  for (int t = 0; t < TT; ++t) {
    // --- 1. Wh = W_hid @ h ; wave wv owns rows [wv*32, wv*32+32) ----------
    for (int r = 0; r < 32; ++r) {
      const int j = wv * 32 + r;
      const float* row = W_hid + (size_t)j * HD;
      float p = 0.f;
      for (int i = l; i < HD; i += 32) p = fmaf(row[i], h_s[i], p);
      p = wave_sum(p);
      if (l == 0) Wh[j] = p;
    }
    __syncthreads();

    // --- 2. scores[s] = w_comb . tanh(enc_proj[s] + Wh) -------------------
    for (int q = 0; q < 16; ++q) {
      const int s = wv * 16 + q;
      const float* ep = enc_proj + (size_t)s * HD;
      float p = 0.f;
      for (int i = l; i < HD; i += 32)
        p = fmaf(w_comb[i], tanhf(ep[i] + Wh[i]), p);
      p = wave_sum(p);
      if (l == 0) sc[s] = p;
    }
    __syncthreads();

    // --- 3. softmax over sc[0..SEQ) ---------------------------------------
    float mv = (tid < SEQ) ? sc[tid] : -3.4e38f;
    mv = wave_max(mv);
    if (l == 0) red[wv] = mv;
    __syncthreads();
    if (wv == 0) {
      float x = wave_max(red[l]);
      if (l == 0) red[0] = x;
    }
    __syncthreads();
    const float mx = red[0];
    __syncthreads();
    float e = (tid < SEQ) ? expf(sc[tid] - mx) : 0.f;
    float sv = wave_sum(e);
    if (l == 0) red[wv] = sv;
    __syncthreads();
    if (wv == 0) {
      float x = wave_sum(red[l]);
      if (l == 0) red[0] = x;
    }
    __syncthreads();
    const float inv = 1.f / red[0];
    __syncthreads();
    if (tid < SEQ) sc[tid] = e * inv;
    __syncthreads();

    // --- 4. ctx = attn @ enc  (coalesced: lane-adjacent h-index) ----------
    {
      float p = 0.f;
      for (int s = 0; s < SEQ; ++s)
        p = fmaf(sc[s], enc[(size_t)s * HD + tid], p);
      ctx[tid] = p;
    }
    __syncthreads();

    // --- 5. gates = emb_pre + W_ih[:,H:]@ctx + W_hh@h ---------------------
    for (int r = 0; r < 128; ++r) {
      const int j = wv * 128 + r;
      const float* wih = W_ih + (size_t)j * (2 * HD) + HD;  // ctx half
      const float* whh = W_hh + (size_t)j * HD;
      float p = 0.f;
      for (int i = l; i < HD; i += 32) p = fmaf(wih[i], ctx[i], p);
      for (int i = l; i < HD; i += 32) p = fmaf(whh[i], h_s[i], p);
      p = wave_sum(p);
      if (l == 0) gates[j] = emb_pre[(size_t)t * (4 * HD) + j] + p;
    }
    __syncthreads();

    // --- 6. LSTM pointwise (torch gate order i, f, g, o) ------------------
    const float ig = gates[tid];
    const float fg = gates[HD + tid];
    const float gg = gates[2 * HD + tid];
    const float og = gates[3 * HD + tid];
    const float si = 1.f / (1.f + expf(-ig));
    const float sf = 1.f / (1.f + expf(-fg));
    const float so = 1.f / (1.f + expf(-og));
    c = sf * c + si * tanhf(gg);
    const float hn = so * tanhf(c);
    __syncthreads();
    h_s[tid] = hn;
    hA[(size_t)t * HD + tid] = (__bf16)hn;
    __syncthreads();
  }
}

// ---------------------------------------------------------------------------
// In-place log_softmax per row of D [TT, N].
// ---------------------------------------------------------------------------
__global__ __launch_bounds__(1024)
void logsoftmax_rows(float* __restrict__ D, int N) {
  __shared__ float red[32];
  float* row = D + (size_t)blockIdx.x * N;
  const int tid = threadIdx.x;
  const int l   = tid & 31;
  const int wv  = tid >> 5;

  float mv = -3.4e38f;
  for (int i = tid; i < N; i += blockDim.x) mv = fmaxf(mv, row[i]);
  mv = wave_max(mv);
  if (l == 0) red[wv] = mv;
  __syncthreads();
  if (wv == 0) {
    float x = wave_max(red[l]);
    if (l == 0) red[0] = x;
  }
  __syncthreads();
  const float mx = red[0];
  __syncthreads();

  float sv = 0.f;
  for (int i = tid; i < N; i += blockDim.x) sv += expf(row[i] - mx);
  sv = wave_sum(sv);
  if (l == 0) red[wv] = sv;
  __syncthreads();
  if (wv == 0) {
    float x = wave_sum(red[l]);
    if (l == 0) red[0] = x;
  }
  __syncthreads();
  const float lse = mx + logf(red[0]);
  for (int i = tid; i < N; i += blockDim.x) row[i] -= lse;
}

// ---------------------------------------------------------------------------
// Host-side orchestration.
// ---------------------------------------------------------------------------
extern "C" void kernel_launch(void* const* d_in, const int* in_sizes, int n_in,
                              void* d_out, int out_size, void* d_ws,
                              size_t ws_size, hipStream_t stream) {
  const int*   start_token = (const int*)  d_in[0];
  // d_in[1] = end_token (never triggers early break in reference) - unused
  const float* h0     = (const float*)d_in[2];
  const float* c0     = (const float*)d_in[3];
  const int*   target = (const int*)  d_in[4];
  const float* enc    = (const float*)d_in[5];   // [1,SEQ,HD]
  const float* embed  = (const float*)d_in[6];   // [VOC,HD]
  const float* W_enc  = (const float*)d_in[7];   // [HD,HD]
  const float* W_hid  = (const float*)d_in[8];   // [HD,HD]
  const float* w_comb = (const float*)d_in[9];   // [HD]
  const float* W_ih   = (const float*)d_in[10];  // [4HD,2HD]
  const float* W_hh   = (const float*)d_in[11];  // [4HD,HD]
  const float* b_ih   = (const float*)d_in[12];  // [4HD]
  const float* b_hh   = (const float*)d_in[13];  // [4HD]
  const float* cls_W  = (const float*)d_in[14];  // [VOC,HD]
  const float* cls_b  = (const float*)d_in[15];  // [VOC]
  float* out = (float*)d_out;                    // [TT,VOC]

  char* ws = (char*)d_ws;
  __bf16* embA = (__bf16*)ws;  ws += (size_t)TT  * HD * sizeof(__bf16);
  __bf16* encA = (__bf16*)ws;  ws += (size_t)SEQ * HD * sizeof(__bf16);
  __bf16* hA   = (__bf16*)ws;  ws += (size_t)TT  * HD * sizeof(__bf16);
  float* emb_pre  = (float*)ws; ws += (size_t)TT  * 4 * HD * sizeof(float);
  float* enc_proj = (float*)ws; ws += (size_t)SEQ * HD * sizeof(float);

  // 1. token gather + fp32->bf16 staging
  gather_embed_bf16<<<TT, 256, 0, stream>>>(start_token, target, embed, embA);
  f32_to_bf16<<<(SEQ * HD + 255) / 256, 256, 0, stream>>>(enc, encA, SEQ * HD);

  // 2. enc_proj[s,j] = sum_k enc[s,k] * W_enc[j,k]   (M=SEQ, N=HD, K=HD)
  gemm_wmma_bf16<<<dim3(HD / 256, SEQ / 16), 256, 0, stream>>>(
      encA, W_enc, HD, nullptr, nullptr, enc_proj, SEQ, HD, HD);

  // 3. emb_pre[t,j] = sum_k W_ih[j,k]*emb[t,k] + b_ih[j] + b_hh[j]
  //    (M=TT, N=4HD, K=HD; W_ih row stride = 2HD, left half only)
  gemm_wmma_bf16<<<dim3(4 * HD / 256, TT / 16), 256, 0, stream>>>(
      embA, W_ih, 2 * HD, b_ih, b_hh, emb_pre, TT, 4 * HD, HD);

  // 4. sequential attention + LSTM recurrence (latency-bound)
  decoder_recurrence<<<1, 1024, 0, stream>>>(
      h0, c0, enc, enc_proj, w_comb, W_hid, W_ih, W_hh, emb_pre, hA);

  // 5. logits[t,v] = sum_k h[t,k]*cls_W[v,k] + cls_b[v]  (dominant GEMM)
  gemm_wmma_bf16<<<dim3(VOC / 256, TT / 16), 256, 0, stream>>>(
      hA, cls_W, HD, cls_b, nullptr, out, TT, VOC, HD);

  // 6. in-place log_softmax per time step
  logsoftmax_rows<<<TT, 1024, 0, stream>>>(out, VOC);
}
